// INTagger_20246475833831
// MI455X (gfx1250) — matrix operand order repaired and checked by reading, not compiled, not run on hardware
//
#include <hip/hip_runtime.h>

// ---------------------------------------------------------------------------
// Types for CDNA5 WMMA / TDM
// ---------------------------------------------------------------------------
typedef __attribute__((ext_vector_type(16))) __bf16 v16bf;
typedef __attribute__((ext_vector_type(8)))  float  v8f;
typedef __attribute__((ext_vector_type(8)))  unsigned int v8u;
typedef __attribute__((ext_vector_type(4)))  unsigned int v4u;
typedef __attribute__((ext_vector_type(4)))  int v4i;
typedef __attribute__((ext_vector_type(8)))  int v8i;

#if defined(__has_builtin)
#  if __has_builtin(__builtin_amdgcn_tensor_load_to_lds)
#    define HAVE_TDM 1
#  endif
#endif
#ifndef HAVE_TDM
#  define HAVE_TDM 0
#endif

__device__ __forceinline__ float bf2f(unsigned short h) {
    unsigned int u = ((unsigned int)h) << 16;
    return __builtin_bit_cast(float, u);
}
__device__ __forceinline__ unsigned short f2bf(float f) {
    unsigned int u = __builtin_bit_cast(unsigned int, f);
    unsigned int r = u + 0x7FFFu + ((u >> 16) & 1u);   // round-to-nearest-even
    return (unsigned short)(r >> 16);
}

// ---------------------------------------------------------------------------
// Problem constants
// ---------------------------------------------------------------------------
#define BB   2048
#define NN   30
#define NVV  14
#define PP   30
#define SS   14
#define NRR  870     // N*(N-1)
#define NTT  420     // N*NV
#define DEE  20
#define DOO  24
#define NCLS 10

// GEMM staging modes
#define MODE_RAW 0   // A tile is final bf16: TDM-copy to LDS
#define MODE_BN  1   // A is previous-layer z: apply BN scale/shift + ReLU
#define MODE_GPP 2   // fused pair-pair gather from xn slabs (TDM)
#define MODE_GPV 3   // fused particle-vertex gather from xn+yn slabs (TDM)

// ---------------------------------------------------------------------------
// TDM helper: 2D tile load Global -> LDS
//   dsize_code: 0=1B,1=2B,2=4B ; dims/strides in elements of that size
// ---------------------------------------------------------------------------
__device__ __forceinline__ void tdm_load_2d(unsigned int lds_off, const void* gptr,
                                            int dsize_code,
                                            unsigned int dim0, unsigned int dim1,
                                            unsigned int stride0,
                                            unsigned int tile0, unsigned int tile1) {
#if HAVE_TDM
    unsigned long long ga = (unsigned long long)gptr;
    v4u g0;
    g0[0] = 1u;                                     // count=1
    g0[1] = lds_off;                                // lds_addr
    g0[2] = (unsigned int)(ga & 0xFFFFFFFFull);     // global_addr lo
    g0[3] = (unsigned int)((ga >> 32) & 0x1FFFFFFull) | (2u << 30);  // hi | type=2
    v8i g1;
    g1[0] = dsize_code << 16;                       // data_size
    g1[1] = (int)((dim0 & 0xFFFFu) << 16);          // tensor_dim0[15:0] @63:48
    g1[2] = (int)(((dim0 >> 16) & 0xFFFFu) | ((dim1 & 0xFFFFu) << 16));
    g1[3] = (int)(((dim1 >> 16) & 0xFFFFu) | ((tile0 & 0xFFFFu) << 16));
    g1[4] = (int)(tile1 & 0xFFFFu);                 // tile_dim1 (tile_dim2=0)
    g1[5] = (int)stride0;                           // tensor_dim0_stride lo32
    g1[6] = 0;
    g1[7] = 0;
    v4i z4 = {0, 0, 0, 0};
#if __clang_major__ >= 23
    v8i z8 = {0, 0, 0, 0, 0, 0, 0, 0};
    __builtin_amdgcn_tensor_load_to_lds(g0, g1, z4, z4, z8, 0);
#else
    __builtin_amdgcn_tensor_load_to_lds(g0, g1, z4, z4, 0);
#endif
#else
    (void)lds_off; (void)gptr; (void)dsize_code; (void)dim0; (void)dim1;
    (void)stride0; (void)tile0; (void)tile1;
#endif
}

// ===========================================================================
// BatchNorm over (B, C, N): per-channel stats -> scale/shift
// ===========================================================================
__global__ void bn3d_stats(const float* __restrict__ X, int C, int Nd,
                           const float* __restrict__ g, const float* __restrict__ beta,
                           float* __restrict__ ss /* [2*C] */) {
    int ch = blockIdx.x;
    int tid = threadIdx.x;
    long total = (long)BB * Nd;
    float s1 = 0.f, s2 = 0.f;
    for (long i = tid; i < total; i += blockDim.x) {
        long b = i / Nd; int n = (int)(i - b * Nd);
        float v = X[(b * C + ch) * Nd + n];
        s1 += v; s2 += v * v;
    }
    __shared__ float r1[256], r2[256];
    r1[tid] = s1; r2[tid] = s2;
    __syncthreads();
    for (int s = 128; s > 0; s >>= 1) {
        if (tid < s) { r1[tid] += r1[tid + s]; r2[tid] += r2[tid + s]; }
        __syncthreads();
    }
    if (tid == 0) {
        float inv = 1.0f / (float)total;
        float mu = r1[0] * inv;
        float var = r2[0] * inv - mu * mu; if (var < 0.f) var = 0.f;
        float sc = g[ch] * rsqrtf(var + 1e-5f);
        ss[ch] = sc;
        ss[C + ch] = beta[ch] - mu * sc;
    }
}

__global__ void bn3d_apply(const float* __restrict__ X, float* __restrict__ Y,
                           const float* __restrict__ ss, int C, int Nd, long total) {
    for (long i = (long)blockIdx.x * blockDim.x + threadIdx.x; i < total;
         i += (long)gridDim.x * blockDim.x) {
        int ch = (int)((i / Nd) % C);
        Y[i] = X[i] * ss[ch] + ss[C + ch];
    }
}

// ===========================================================================
// Weight packing into per-lane WMMA B-fragment order
// ===========================================================================
__global__ void pack_w_frag(const float* __restrict__ W, const float* __restrict__ bias,
                            int dout, int din, int KP, int NP,
                            unsigned int* __restrict__ Wf, float* __restrict__ biasp) {
    int NKS = KP / 32;
    int total = (NP / 16) * NKS * 32 * 8;
    int i = blockIdx.x * blockDim.x + threadIdx.x;
    if (i < total) {
        int v = i & 7;
        int lane = (i >> 3) & 31;
        int frag = i >> 8;               // nt*NKS + ks
        int ks = frag % NKS, nt = frag / NKS;
        int n = nt * 16 + (lane & 15);
        int k = ks * 32 + ((v >> 2) << 4) + ((lane >> 4) << 3) + 2 * (v & 3);
        float lo = (k     < din && n < dout) ? W[n * din + k]     : 0.f;
        float hi = (k + 1 < din && n < dout) ? W[n * din + k + 1] : 0.f;
        Wf[i] = (unsigned int)f2bf(lo) | ((unsigned int)f2bf(hi) << 16);
    }
    if (i < NP) biasp[i] = (i < dout) ? bias[i] : 0.0f;
}

// ===========================================================================
// Gather for fo (small: 61440 x 96)
// ===========================================================================
__global__ void gather_fo(const float* __restrict__ xn, const float* __restrict__ Epp,
                          const float* __restrict__ Epv, unsigned short* __restrict__ A,
                          long total) {
    for (long idx = (long)blockIdx.x * blockDim.x + threadIdx.x; idx < total;
         idx += (long)gridDim.x * blockDim.x) {
        long row = idx / 96; int k = (int)(idx - row * 96);
        int b = (int)(row / NN); int n = (int)(row - (long)b * NN);
        float v = 0.0f;
        if (k < PP)               v = xn[((long)b * PP + k) * NN + n];
        else if (k < PP + DEE)    v = Epp[((long)b * DEE + (k - PP)) * NN + n];
        else if (k < PP + 2*DEE)  v = Epv[((long)b * DEE + (k - PP - DEE)) * NN + n];
        A[idx] = f2bf(v);
    }
}

// ===========================================================================
// Core WMMA GEMM (templated on KP, NP, MODE):
//   Z[M,NP] = act(A)[M,KP] @ W[KP,NP] + bias ; per-block column stats -> pstats
//   Block tile: 32 rows x NP cols; waves = 2*(NP/16); blockDim = 64*(NP/16).
// ===========================================================================
template<int KP, int NP, int MODE>
__global__ __launch_bounds__(256)
void gemm_wmma(const unsigned short* __restrict__ A,
               const unsigned int* __restrict__ Wf,
               const float* __restrict__ biasp,
               const float* __restrict__ ss,
               const float* __restrict__ X0,
               const float* __restrict__ X1,
               unsigned short* __restrict__ Z,
               float* __restrict__ pstats,
               long M) {
    constexpr int NT  = NP / 16;
    constexpr int NKS = KP / 32;
    __shared__ __align__(16) unsigned short sA[32 * KP];
    __shared__ float sStat[2 * NP];
    __shared__ float xnS[(MODE == MODE_GPP || MODE == MODE_GPV) ? 2 * PP * NN : 1];
    __shared__ float ynS[(MODE == MODE_GPV) ? 2 * SS * NVV : 1];

    const int tid = threadIdx.x;
    const int nthreads = blockDim.x;
    const long rowbase = (long)blockIdx.x * 32;

    for (int i = tid; i < 2 * NP; i += nthreads) sStat[i] = 0.f;

    if (MODE == MODE_BN) {
        // Stage A tile with previous-layer BN + ReLU folded in.
        for (int i = tid; i < 32 * KP; i += nthreads) {
            int k = i % KP;
            float v = bf2f(A[rowbase * KP + i]);
            v = v * ss[k] + ss[KP + k];
            sA[i] = f2bf(v > 0.f ? v : 0.f);
        }
        __syncthreads();
    } else if (MODE == MODE_RAW) {
#if HAVE_TDM
        if (tid == 0) {
            tdm_load_2d((unsigned int)(unsigned long long)(const void*)&sA[0],
                        A + rowbase * KP, /*2B*/1,
                        KP, (unsigned int)(M - rowbase), KP, KP, 32);
        }
        __builtin_amdgcn_s_wait_tensorcnt(0);
#else
        for (int i = tid; i < 32 * KP / 8; i += nthreads)
            ((uint4*)sA)[i] = *(const uint4*)(A + rowbase * KP + (long)i * 8);
#endif
        __syncthreads();
    } else if (MODE == MODE_GPP) {
        const int b0 = (int)(rowbase / NRR);
#if HAVE_TDM
        if (tid == 0) {
            tdm_load_2d((unsigned int)(unsigned long long)(const void*)&xnS[0],
                        X0 + (long)b0 * (PP * NN), /*4B*/2,
                        PP * NN, (unsigned int)(BB - b0), PP * NN, PP * NN, 2);
        }
        __builtin_amdgcn_s_wait_tensorcnt(0);
#else
        for (int i = tid; i < 2 * PP * NN; i += nthreads)
            xnS[i] = X0[(long)b0 * (PP * NN) + i];
#endif
        __syncthreads();
        for (int i = tid; i < 32 * KP; i += nthreads) {
            int rr = i / KP, k = i - rr * KP;
            long grow = rowbase + rr;
            int b = (int)(grow / NRR);
            int e = (int)(grow - (long)b * NRR);
            int loc = (b - b0) * (PP * NN);
            int rI = e / 29; int j = e - 29 * rI;
            int sI = j + (j >= rI ? 1 : 0);
            float v = 0.f;
            if (k < PP)          v = xnS[loc + k * NN + rI];
            else if (k < 2 * PP) v = xnS[loc + (k - PP) * NN + sI];
            sA[i] = f2bf(v);
        }
        __syncthreads();
    } else { // MODE_GPV
        const int b0 = (int)(rowbase / NTT);
#if HAVE_TDM
        if (tid == 0) {
            tdm_load_2d((unsigned int)(unsigned long long)(const void*)&xnS[0],
                        X0 + (long)b0 * (PP * NN), 2,
                        PP * NN, (unsigned int)(BB - b0), PP * NN, PP * NN, 2);
            tdm_load_2d((unsigned int)(unsigned long long)(const void*)&ynS[0],
                        X1 + (long)b0 * (SS * NVV), 2,
                        SS * NVV, (unsigned int)(BB - b0), SS * NVV, SS * NVV, 2);
        }
        __builtin_amdgcn_s_wait_tensorcnt(0);
#else
        for (int i = tid; i < 2 * PP * NN; i += nthreads)
            xnS[i] = X0[(long)b0 * (PP * NN) + i];
        for (int i = tid; i < 2 * SS * NVV; i += nthreads)
            ynS[i] = X1[(long)b0 * (SS * NVV) + i];
#endif
        __syncthreads();
        for (int i = tid; i < 32 * KP; i += nthreads) {
            int rr = i / KP, k = i - rr * KP;
            long grow = rowbase + rr;
            int b = (int)(grow / NTT);
            int e = (int)(grow - (long)b * NTT);
            int kc = e / NVV; int vc = e - NVV * kc;
            float v = 0.f;
            if (k < PP)           v = xnS[(b - b0) * (PP * NN) + k * NN + kc];
            else if (k < PP + SS) v = ynS[(b - b0) * (SS * NVV) + (k - PP) * NVV + vc];
            sA[i] = f2bf(v);
        }
        __syncthreads();
    }

    const int wave = tid >> 5, lane = tid & 31;
    const int mt = wave / NT, nt = wave - mt * NT;
    const int l15 = lane & 15, lhi = lane >> 4;
    const int n = nt * 16 + l15;

    v8f c;
    {
        float bv = biasp[n];
        #pragma unroll
        for (int v = 0; v < 8; ++v) c[v] = bv;
    }

    const uint4* __restrict__ bbase =
        (const uint4*)Wf + ((long)(nt * NKS) * 32 + lane) * 2;

    #pragma unroll
    for (int ks = 0; ks < NKS; ++ks) {
        const unsigned short* ap = &sA[(mt * 16 + l15) * KP + ks * 32 + (lhi << 3)];
        uint4 alo = *(const uint4*)(ap);
        uint4 ahi = *(const uint4*)(ap + 16);
        v8u a32 = {alo.x, alo.y, alo.z, alo.w, ahi.x, ahi.y, ahi.z, ahi.w};
        const uint4* bp = bbase + (long)ks * 64;
        uint4 b0v = bp[0];
        uint4 b1v = bp[1];
        v8u b32 = {b0v.x, b0v.y, b0v.z, b0v.w, b1v.x, b1v.y, b1v.z, b1v.w};
        v16bf av = __builtin_bit_cast(v16bf, a32);
        v16bf bv = __builtin_bit_cast(v16bf, b32);
        c = __builtin_amdgcn_wmma_f32_16x16x32_bf16(false, av, false, bv,
                                                    (short)0, c, false, false);
    }

    // Store C tile (pre-BN z, bf16). VGPR v -> row v (lanes 0-15) / v+8.
    const long rb = rowbase + mt * 16 + (lhi << 3);
    #pragma unroll
    for (int v = 0; v < 8; ++v) {
        Z[(rb + v) * NP + n] = f2bf(c[v]);
    }

    // Fused per-block column stats (sum, sumsq).
    float s1 = 0.f, s2 = 0.f;
    #pragma unroll
    for (int v = 0; v < 8; ++v) { float zv = c[v]; s1 += zv; s2 += zv * zv; }
    s1 += __shfl_xor(s1, 16);
    s2 += __shfl_xor(s2, 16);
    if (lane < 16) {
        atomicAdd(&sStat[n], s1);
        atomicAdd(&sStat[NP + n], s2);
    }
    __syncthreads();
    for (int i = tid; i < 2 * NP; i += nthreads)
        pstats[(long)i * gridDim.x + blockIdx.x] = sStat[i];
}

// ===========================================================================
// Stats reduction: pstats [2*NP][nblocks] -> red[2*NP]
// ===========================================================================
__global__ void reduce_stats(const float* __restrict__ pstats, long nblocks,
                             float* __restrict__ red) {
    int c = blockIdx.x;
    const float* p = pstats + (long)c * nblocks;
    float s = 0.f;
    for (long i = threadIdx.x; i < nblocks; i += blockDim.x) s += p[i];
    __shared__ float r[256];
    r[threadIdx.x] = s;
    __syncthreads();
    for (int st = 128; st > 0; st >>= 1) {
        if (threadIdx.x < st) r[threadIdx.x] += r[threadIdx.x + st];
        __syncthreads();
    }
    if (threadIdx.x == 0) red[c] = r[0];
}

__global__ void finalize_bn(const float* __restrict__ red, int NP, long M,
                            const float* __restrict__ g, const float* __restrict__ beta,
                            int dout, float* __restrict__ ss /* 2*NP */) {
    int c = threadIdx.x;
    if (c >= NP) return;
    float inv = 1.0f / (float)M;
    float mu = red[c] * inv;
    float var = red[NP + c] * inv - mu * mu; if (var < 0.f) var = 0.f;
    float rstd = rsqrtf(var + 1e-5f);
    float gg = (c < dout) ? g[c] : 0.0f;
    float bb = (c < dout) ? beta[c] : 0.0f;
    float sc = gg * rstd;
    ss[c] = sc;
    ss[NP + c] = bb - mu * sc;
}

// ===========================================================================
// Scatter-sums (segment sums by receiver index), per-batch LDS accumulators
// ===========================================================================
__global__ void scatter_pp(const unsigned short* __restrict__ Z3,
                           const float* __restrict__ ss,
                           float* __restrict__ Ebar /* B x DE x N */) {
    int b = blockIdx.x;
    __shared__ float acc[NN * DEE];
    for (int i = threadIdx.x; i < NN * DEE; i += blockDim.x) acc[i] = 0.f;
    __syncthreads();
    for (int i = threadIdx.x; i < NRR * DEE; i += blockDim.x) {
        int e = i / DEE, ch = i - e * DEE;
        int r = e / 29;
        float v = bf2f(Z3[((long)b * NRR + e) * 32 + ch]);
        v = v * ss[ch] + ss[32 + ch];
        v = v > 0.f ? v : 0.f;
        atomicAdd(&acc[r * DEE + ch], v);
    }
    __syncthreads();
    for (int i = threadIdx.x; i < NN * DEE; i += blockDim.x) {
        int nn = i / DEE, ch = i - nn * DEE;
        Ebar[((long)b * DEE + ch) * NN + nn] = acc[i];
    }
}

__global__ void scatter_pv(const unsigned short* __restrict__ Z3,
                           const float* __restrict__ ss,
                           float* __restrict__ Ebar) {
    int b = blockIdx.x;
    __shared__ float acc[NN * DEE];
    for (int i = threadIdx.x; i < NN * DEE; i += blockDim.x) acc[i] = 0.f;
    __syncthreads();
    for (int i = threadIdx.x; i < NTT * DEE; i += blockDim.x) {
        int e = i / DEE, ch = i - e * DEE;
        int kidx = e / NVV;
        float v = bf2f(Z3[((long)b * NTT + e) * 32 + ch]);
        v = v * ss[ch] + ss[32 + ch];
        v = v > 0.f ? v : 0.f;
        atomicAdd(&acc[kidx * DEE + ch], v);
    }
    __syncthreads();
    for (int i = threadIdx.x; i < NN * DEE; i += blockDim.x) {
        int nn = i / DEE, ch = i - nn * DEE;
        Ebar[((long)b * DEE + ch) * NN + nn] = acc[i];
    }
}

// ===========================================================================
// Head: BN+ReLU on z3_o, pool over N, FC 24->10
// ===========================================================================
__global__ void head_kernel(const unsigned short* __restrict__ Z3,
                            const float* __restrict__ ss,
                            const float* __restrict__ Wfc,
                            const float* __restrict__ bfc,
                            float* __restrict__ out) {
    int b = blockIdx.x;
    int tid = threadIdx.x;   // 32
    __shared__ float pooled[DOO];
    if (tid < DOO) {
        float s = 0.f;
        for (int n = 0; n < NN; ++n) {
            float v = bf2f(Z3[((long)b * NN + n) * 32 + tid]);
            v = v * ss[tid] + ss[32 + tid];
            s += v > 0.f ? v : 0.f;
        }
        pooled[tid] = s;
    }
    __syncthreads();
    if (tid < NCLS) {
        float s = bfc[tid];
        for (int c = 0; c < DOO; ++c) s += pooled[c] * Wfc[tid * DOO + c];
        out[b * NCLS + tid] = s;
    }
}

// ===========================================================================
// Host orchestration
// ===========================================================================
struct LayerCfg { int din, dout, KP, NP; };

extern "C" void kernel_launch(void* const* d_in, const int* in_sizes, int n_in,
                              void* d_out, int out_size, void* d_ws, size_t ws_size,
                              hipStream_t stream) {
    (void)in_sizes; (void)n_in; (void)out_size; (void)ws_size;

    const float* x = (const float*)d_in[0];
    const float* y = (const float*)d_in[1];
    const float* bnx_g = (const float*)d_in[2];
    const float* bnx_b = (const float*)d_in[3];
    const float* bny_g = (const float*)d_in[4];
    const float* bny_b = (const float*)d_in[5];
    const float* Wfc = (const float*)d_in[42];
    const float* bfc = (const float*)d_in[43];
    float* out = (float*)d_out;

    // ---- workspace carve ----
    char* w = (char*)d_ws;
    auto alloc = [&](size_t bytes) -> void* {
        void* p = (void*)w;
        w += (bytes + 255) & ~(size_t)255;
        return p;
    };
    float* xn  = (float*)alloc((size_t)BB * PP * NN * 4);
    float* yn  = (float*)alloc((size_t)BB * SS * NVV * 4);
    float* ssx = (float*)alloc(2 * PP * 4);
    float* ssy = (float*)alloc(2 * SS * 4);

    static const LayerCfg cfg[9] = {
        {60, 60, 64, 64}, {60, 60, 64, 64}, {60, 20, 64, 32},   // fr
        {44, 60, 64, 64}, {60, 60, 64, 64}, {60, 20, 64, 32},   // fr_pv
        {70, 60, 96, 64}, {60, 60, 64, 64}, {60, 24, 64, 32},   // fo
    };
    unsigned int* Wf[9];
    float* biasp[9];
    float* ssz[9];
    for (int l = 0; l < 9; ++l) {
        Wf[l]    = (unsigned int*)alloc(16384);
        biasp[l] = (float*)alloc((size_t)cfg[l].NP * 4);
        ssz[l]   = (float*)alloc((size_t)2 * cfg[l].NP * 4);
    }
    const long M_pp = (long)BB * NRR;   // 1,781,760
    const long M_pv = (long)BB * NTT;   //   860,160
    const long M_fo = (long)BB * NN;    //    61,440
    float* pstats = (float*)alloc((size_t)2 * 64 * (M_pp / 32) * 4);   // 28.5 MB
    float* red    = (float*)alloc(2 * 64 * 4);
    float* Epp = (float*)alloc((size_t)BB * DEE * NN * 4);
    float* Epv = (float*)alloc((size_t)BB * DEE * NN * 4);
    unsigned short* bufA = (unsigned short*)alloc((size_t)M_pp * 96 * 2);
    unsigned short* bufB = (unsigned short*)alloc((size_t)M_pp * 64 * 2);

    // ---- input batchnorms ----
    bn3d_stats<<<PP, 256, 0, stream>>>(x, PP, NN, bnx_g, bnx_b, ssx);
    bn3d_stats<<<SS, 256, 0, stream>>>(y, SS, NVV, bny_g, bny_b, ssy);
    {
        long tx = (long)BB * PP * NN, ty = (long)BB * SS * NVV;
        bn3d_apply<<<(unsigned)((tx + 255) / 256), 256, 0, stream>>>(x, xn, ssx, PP, NN, tx);
        bn3d_apply<<<(unsigned)((ty + 255) / 256), 256, 0, stream>>>(y, yn, ssy, SS, NVV, ty);
    }

    // ---- pack all 9 MLP layers into fragment order ----
    for (int l = 0; l < 9; ++l) {
        int base = 6 + (l / 3) * 12 + (l % 3) * 4;
        int nelem = (cfg[l].NP / 16) * (cfg[l].KP / 32) * 32 * 8;
        pack_w_frag<<<(nelem + 255) / 256, 256, 0, stream>>>(
            (const float*)d_in[base], (const float*)d_in[base + 1],
            cfg[l].dout, cfg[l].din, cfg[l].KP, cfg[l].NP, Wf[l], biasp[l]);
    }

    // stats pipeline after a GEMM of layer l over M rows
    auto stats = [&](int l, long M) {
        int base = 6 + (l / 3) * 12 + (l % 3) * 4;
        long nblocks = M / 32;
        reduce_stats<<<2 * cfg[l].NP, 256, 0, stream>>>(pstats, nblocks, red);
        finalize_bn<<<1, 64, 0, stream>>>(red, cfg[l].NP, M,
                                          (const float*)d_in[base + 2],
                                          (const float*)d_in[base + 3],
                                          cfg[l].dout, ssz[l]);
    };

    // ---- fr: pair-pair branch (fused gather layer0) ----
    {
        unsigned g0 = (unsigned)(M_pp / 32);
        gemm_wmma<64, 64, MODE_GPP><<<g0, 256, 0, stream>>>(
            nullptr, Wf[0], biasp[0], nullptr, xn, nullptr, bufA, pstats, M_pp);
        stats(0, M_pp);
        gemm_wmma<64, 64, MODE_BN><<<g0, 256, 0, stream>>>(
            bufA, Wf[1], biasp[1], ssz[0], nullptr, nullptr, bufB, pstats, M_pp);
        stats(1, M_pp);
        gemm_wmma<64, 32, MODE_BN><<<g0, 128, 0, stream>>>(
            bufB, Wf[2], biasp[2], ssz[1], nullptr, nullptr, bufA, pstats, M_pp);
        stats(2, M_pp);
        scatter_pp<<<BB, 256, 0, stream>>>(bufA, ssz[2], Epp);
    }
    // ---- fr_pv: particle-vertex branch (fused gather layer0) ----
    {
        unsigned g0 = (unsigned)(M_pv / 32);
        gemm_wmma<64, 64, MODE_GPV><<<g0, 256, 0, stream>>>(
            nullptr, Wf[3], biasp[3], nullptr, xn, yn, bufA, pstats, M_pv);
        stats(3, M_pv);
        gemm_wmma<64, 64, MODE_BN><<<g0, 256, 0, stream>>>(
            bufA, Wf[4], biasp[4], ssz[3], nullptr, nullptr, bufB, pstats, M_pv);
        stats(4, M_pv);
        gemm_wmma<64, 32, MODE_BN><<<g0, 128, 0, stream>>>(
            bufB, Wf[5], biasp[5], ssz[4], nullptr, nullptr, bufA, pstats, M_pv);
        stats(5, M_pv);
        scatter_pv<<<BB, 256, 0, stream>>>(bufA, ssz[5], Epv);
    }
    // ---- fo: object network + head ----
    {
        long total = M_fo * 96;
        gather_fo<<<(unsigned)((total + 255) / 256), 256, 0, stream>>>(xn, Epp, Epv, bufA, total);
        unsigned g0 = (unsigned)(M_fo / 32);
        gemm_wmma<96, 64, MODE_RAW><<<g0, 256, 0, stream>>>(
            bufA, Wf[6], biasp[6], nullptr, nullptr, nullptr, bufB, pstats, M_fo);
        stats(6, M_fo);
        gemm_wmma<64, 64, MODE_BN><<<g0, 256, 0, stream>>>(
            bufB, Wf[7], biasp[7], ssz[6], nullptr, nullptr, bufA, pstats, M_fo);
        stats(7, M_fo);
        gemm_wmma<64, 32, MODE_BN><<<g0, 128, 0, stream>>>(
            bufA, Wf[8], biasp[8], ssz[7], nullptr, nullptr, bufB, pstats, M_fo);
        stats(8, M_fo);
        head_kernel<<<BB, 32, 0, stream>>>(bufB, ssz[8], Wfc, bfc, out);
    }
}
